// MultiHeadAttention_78692390797840
// MI455X (gfx1250) — matrix operand order repaired and checked by reading, not compiled, not run on hardware
//
#include <hip/hip_runtime.h>

// ---------------------------------------------------------------------------
// MHA forward for MI455X (gfx1250), wave32, WMMA bf16, async LDS staging.
// B=4, S=2048, D=1024, H=8, HD=128.
// ---------------------------------------------------------------------------

typedef __attribute__((ext_vector_type(16))) __bf16          v16bf;
typedef __attribute__((ext_vector_type(8)))  float           v8f;
typedef __attribute__((ext_vector_type(8)))  unsigned short  v8u;
typedef __attribute__((ext_vector_type(4)))  float           v4f;
typedef __attribute__((ext_vector_type(4)))  unsigned short  v4u;
typedef __attribute__((ext_vector_type(4)))  int             v4i;

union Frag16 {            // 16 bf16 = one WMMA A or B operand (8 VGPRs)
    v8u   h[2];
    v16bf bf;
};

__device__ __forceinline__ unsigned short f2bf(float f) {
    unsigned int x = __builtin_bit_cast(unsigned int, f);
    unsigned int r = x + 0x7FFFu + ((x >> 16) & 1u);   // round-to-nearest-even
    return (unsigned short)(r >> 16);
}

// --- CDNA5 async global->LDS copy of 16 bytes per lane ----------------------
__device__ __forceinline__ void async_cp16(const unsigned short* g,
                                           unsigned short* l) {
#if __has_builtin(__builtin_amdgcn_global_load_async_to_lds_b128)
    __builtin_amdgcn_global_load_async_to_lds_b128(
        (__attribute__((address_space(1))) v4i*)g,
        (__attribute__((address_space(3))) v4i*)l, 0, 0);
#else
    unsigned loff = (unsigned)(unsigned long long)
        (__attribute__((address_space(3))) unsigned short*)l;
    asm volatile("global_load_async_to_lds_b128 %0, %1, off"
                 :: "v"(loff), "v"((unsigned long long)(size_t)g) : "memory");
#endif
}

__device__ __forceinline__ void wait_async0() {
#if __has_builtin(__builtin_amdgcn_s_wait_asynccnt)
    __builtin_amdgcn_s_wait_asynccnt(0);
#else
    asm volatile("s_wait_asynccnt 0x0" ::: "memory");
#endif
}

__device__ __forceinline__ v8f wmma_bf16(const Frag16& a, const Frag16& b,
                                         v8f c) {
    return __builtin_amdgcn_wmma_f32_16x16x32_bf16(
        false, a.bf, false, b.bf, (short)0, c, false, false);
}

// ---------------------------------------------------------------------------
// f32 -> bf16 pack, 4 elements / thread
// ---------------------------------------------------------------------------
__global__ void cvt_bf16_kernel(const float* __restrict__ in,
                                unsigned short* __restrict__ out, int n4) {
    int i = blockIdx.x * blockDim.x + threadIdx.x;
    if (i < n4) {
        v4f x = ((const v4f*)in)[i];
        v4u y;
        #pragma unroll
        for (int k = 0; k < 4; ++k) y[k] = f2bf(x[k]);
        ((v4u*)out)[i] = y;
    }
}

// ---------------------------------------------------------------------------
// Projection GEMM:  C[M=8192, N=1024] = X[M,1024] * W^T   (W is [out,in])
// One wave computes a 32x64 tile (2 A-frags x 4 B-frags, 8 WMMA / k-step),
// register double-buffered so WMMAs overlap the next step's loads.
// mode 0: bf16 [B,H,S,HD]; mode 1: bf16 [B,H,HD,S] (V^T); mode 2: f32 [M,N].
// ---------------------------------------------------------------------------
__global__ __launch_bounds__(128) void proj_wmma_kernel(
    const unsigned short* __restrict__ X, const unsigned short* __restrict__ W,
    unsigned short* __restrict__ outBF, float* __restrict__ outF,
    int mode, float scale) {
    const int lane = threadIdx.x & 31;
    const int wave = threadIdx.x >> 5;
    const int t    = blockIdx.x * 4 + wave;        // 4096 waves
    const int m0   = (t >> 4) * 32;                // 256 row tiles of 32
    const int n0   = (t & 15) * 64;                // 16 col strips of 64
    const int half = lane >> 4;
    const int ln   = lane & 15;

    v8f acc[2][4] = {};
    Frag16 a[2][2], b[2][4];

    auto loadAB = [&](int set, int k0) {
        #pragma unroll
        for (int i = 0; i < 2; ++i) {
            const unsigned short* pa =
                X + (size_t)(m0 + i * 16 + ln) * 1024 + k0 + half * 8;
            a[set][i].h[0] = *(const v8u*)(pa);
            a[set][i].h[1] = *(const v8u*)(pa + 16);
        }
        #pragma unroll
        for (int j = 0; j < 4; ++j) {
            const unsigned short* pb =
                W + (size_t)(n0 + j * 16 + ln) * 1024 + k0 + half * 8;
            b[set][j].h[0] = *(const v8u*)(pb);
            b[set][j].h[1] = *(const v8u*)(pb + 16);
        }
    };
    auto mmaStep = [&](int set) {
        #pragma unroll
        for (int i = 0; i < 2; ++i)
            #pragma unroll
            for (int j = 0; j < 4; ++j)
                acc[i][j] = wmma_bf16(a[set][i], b[set][j], acc[i][j]);
    };

    loadAB(0, 0);
    for (int kc = 0; kc < 32; kc += 2) {           // K=1024, 32 steps of 32
        loadAB(1, (kc + 1) * 32);
        mmaStep(0);
        if (kc + 2 < 32) loadAB(0, (kc + 2) * 32);
        mmaStep(1);
    }

    // C layout: element (m,n): vgpr m%8, lane n + 16*(m/8)
    #pragma unroll
    for (int i = 0; i < 2; ++i)
        #pragma unroll
        for (int j = 0; j < 4; ++j)
            #pragma unroll
            for (int r = 0; r < 8; ++r) {
                float val = acc[i][j][r] * scale;
                int row = m0 + i * 16 + r + 8 * half;   // [0,8192)
                int col = n0 + j * 16 + ln;             // [0,1024)
                if (mode == 2) {
                    outF[(size_t)row * 1024 + col] = val;
                } else {
                    int bb = row >> 11, s = row & 2047;
                    int hh = col >> 7,  d = col & 127;
                    size_t idx = (mode == 0)
                        ? (((size_t)bb * 8 + hh) * 2048 + s) * 128 + d
                        : (((size_t)bb * 8 + hh) * 128 + d) * 2048 + s;
                    outBF[idx] = f2bf(val);
                }
            }
}

// ---------------------------------------------------------------------------
// Flash attention. Block = 4 waves, all on the same (b,h); each wave owns a
// 16-row Q tile. Per 32-key step the block stages K (32x128) and V^T (128x32)
// tiles into double-buffered LDS with async global->LDS copies, overlapped
// with the previous step's WMMAs + online softmax.
// ---------------------------------------------------------------------------
__global__ __launch_bounds__(128) void flash_attn_kernel(
    const unsigned short* __restrict__ Q,   // [B,H,S,HD] bf16 (pre-scaled)
    const unsigned short* __restrict__ K,   // [B,H,S,HD] bf16
    const unsigned short* __restrict__ Vt,  // [B,H,HD,S] bf16
    const float* __restrict__ mask,         // [S,S] f32
    unsigned short* __restrict__ ctx) {     // [B,S,D] bf16 (head-concat)
    __shared__ unsigned short Klds[2][32 * 128];   // 2 x 8 KB
    __shared__ unsigned short Vlds[2][128 * 32];   // 2 x 8 KB
    __shared__ unsigned short Plds[4][16 * 32];    // 4 KB

    const int tid  = threadIdx.x;
    const int lane = tid & 31;
    const int wave = tid >> 5;
    const int bh   = blockIdx.x >> 5;                       // 0..31
    const int q0   = ((blockIdx.x & 31) * 4 + wave) * 16;   // wave's q tile
    const int bb   = bh >> 3, hh = bh & 7;
    const int half = lane >> 4;
    const int ln   = lane & 15;

    const unsigned short* Qb = Q  + ((size_t)bh * 2048 + q0) * 128;
    const unsigned short* Kb = K  + (size_t)bh * 2048 * 128;
    const unsigned short* Vb = Vt + (size_t)bh * 128 * 2048;

    // Cooperative async stage of one 32-key tile (K: 8KB, V: 8KB)
    auto stage = [&](int buf, int k0) {
        #pragma unroll
        for (int tchunk = 0; tchunk < 4; ++tchunk) {
            int chunk = tid + tchunk * 128;        // 0..511, 16B each
            int row = chunk >> 4, c = chunk & 15;  // K tile: [32][128]
            async_cp16(Kb + (size_t)(k0 + row) * 128 + c * 8,
                       &Klds[buf][row * 128 + c * 8]);
        }
        #pragma unroll
        for (int tchunk = 0; tchunk < 4; ++tchunk) {
            int chunk = tid + tchunk * 128;        // V tile: [128][32]
            int d = chunk >> 2, c = chunk & 3;
            async_cp16(Vb + (size_t)d * 2048 + k0 + c * 8,
                       &Vlds[buf][d * 32 + c * 8]);
        }
    };

    // Preload 4 Q A-fragments (HD=128 = 4 chunks of K=32)
    Frag16 qa[4];
    #pragma unroll
    for (int c = 0; c < 4; ++c) {
        const unsigned short* p = Qb + (size_t)ln * 128 + c * 32 + half * 8;
        qa[c].h[0] = *(const v8u*)(p);
        qa[c].h[1] = *(const v8u*)(p + 16);
    }

    float mprev[8], lsum[8];
    v8f o[8] = {};
    #pragma unroll
    for (int r = 0; r < 8; ++r) { mprev[r] = -1e30f; lsum[r] = 0.0f; }

    stage(0, 0);
    for (int kt = 0; kt < 64; ++kt) {
        const int k0  = kt * 32;
        const int buf = kt & 1;

        wait_async0();          // this wave's copies for tile kt are in LDS
        __syncthreads();        // all waves' copies are in LDS; all reads of
                                // the buffer we are about to overwrite retired
        if (kt < 63) stage(buf ^ 1, k0 + 32);

        // ---- scores: two 16x16 C tiles from LDS K tile ----
        v8f s0 = {}, s1 = {};
        #pragma unroll
        for (int c = 0; c < 4; ++c) {
            Frag16 kb0, kb1;
            const unsigned short* p0 = &Klds[buf][ln * 128 + c * 32 + half * 8];
            kb0.h[0] = *(const v8u*)(p0);
            kb0.h[1] = *(const v8u*)(p0 + 16);
            const unsigned short* p1 = p0 + 16 * 128;
            kb1.h[0] = *(const v8u*)(p1);
            kb1.h[1] = *(const v8u*)(p1 + 16);
            s0 = wmma_bf16(qa[c], kb0, s0);
            s1 = wmma_bf16(qa[c], kb1, s1);
        }

        // ---- online softmax (row m = r + 8*half lives in one 16-lane half) --
        #pragma unroll
        for (int r = 0; r < 8; ++r) {
            const int m = r + 8 * half;
            const float* mp = mask + (size_t)(q0 + m) * 2048 + k0 + ln;
            float a0 = s0[r] + mp[0];
            float a1 = s1[r] + mp[16];
            float tmax = fmaxf(a0, a1);
            tmax = fmaxf(tmax, __shfl_xor(tmax, 1, 32));
            tmax = fmaxf(tmax, __shfl_xor(tmax, 2, 32));
            tmax = fmaxf(tmax, __shfl_xor(tmax, 4, 32));
            tmax = fmaxf(tmax, __shfl_xor(tmax, 8, 32));
            float mnew = fmaxf(mprev[r], tmax);
            float p0 = __expf(a0 - mnew);
            float p1 = __expf(a1 - mnew);
            float rs = p0 + p1;
            rs += __shfl_xor(rs, 1, 32);
            rs += __shfl_xor(rs, 2, 32);
            rs += __shfl_xor(rs, 4, 32);
            rs += __shfl_xor(rs, 8, 32);
            float corr = __expf(mprev[r] - mnew);
            lsum[r] = lsum[r] * corr + rs;
            mprev[r] = mnew;
            #pragma unroll
            for (int f = 0; f < 8; ++f) o[f][r] *= corr;
            Plds[wave][m * 32 + ln]      = f2bf(p0);
            Plds[wave][m * 32 + 16 + ln] = f2bf(p1);
        }
        asm volatile("s_wait_dscnt 0x0" ::: "memory");  // wave-private P RAW

        // ---- P A-fragment from LDS (16x32 bf16, exact K=32 fit) ----
        Frag16 pa;
        {
            const unsigned short* p = &Plds[wave][ln * 32 + half * 8];
            pa.h[0] = *(const v8u*)(p);
            pa.h[1] = *(const v8u*)(p + 16);
        }

        // ---- O += P * V from LDS V tile : 8 HD chunks of 16 ----
        #pragma unroll
        for (int f = 0; f < 8; ++f) {
            Frag16 vbf;
            const unsigned short* p = &Vlds[buf][(f * 16 + ln) * 32 + half * 8];
            vbf.h[0] = *(const v8u*)(p);
            vbf.h[1] = *(const v8u*)(p + 16);
            o[f] = wmma_bf16(pa, vbf, o[f]);
        }
    }

    // ---- epilogue: normalize and store context [B,S,D] head-concat ----
    #pragma unroll
    for (int r = 0; r < 8; ++r) {
        float inv = 1.0f / lsum[r];
        const int m = r + 8 * half;
        size_t base = ((size_t)bb * 2048 + q0 + m) * 1024 + hh * 128 + ln;
        #pragma unroll
        for (int f = 0; f < 8; ++f)
            ctx[base + f * 16] = f2bf(o[f][r] * inv);
    }
}

// ---------------------------------------------------------------------------
// Host launcher
// ---------------------------------------------------------------------------
extern "C" void kernel_launch(void* const* d_in, const int* in_sizes, int n_in,
                              void* d_out, int out_size, void* d_ws, size_t ws_size,
                              hipStream_t stream) {
    (void)in_sizes; (void)n_in; (void)out_size; (void)ws_size;
    const float* x    = (const float*)d_in[0];   // [B,S,D]
    const float* mask = (const float*)d_in[1];   // [1,1,S,S]
    const float* wq   = (const float*)d_in[2];
    const float* wk   = (const float*)d_in[3];
    const float* wv   = (const float*)d_in[4];
    const float* wo   = (const float*)d_in[5];
    float* out = (float*)d_out;

    char* ws = (char*)d_ws;
    const size_t MB = 1024 * 1024;
    unsigned short* Xbf = (unsigned short*)(ws + 0);        // 16 MB
    unsigned short* Wqb = (unsigned short*)(ws + 16 * MB);  //  2 MB
    unsigned short* Wkb = (unsigned short*)(ws + 18 * MB);
    unsigned short* Wvb = (unsigned short*)(ws + 20 * MB);
    unsigned short* Wob = (unsigned short*)(ws + 22 * MB);
    unsigned short* Qb  = (unsigned short*)(ws + 24 * MB);  // 16 MB
    unsigned short* Kb  = (unsigned short*)(ws + 40 * MB);  // 16 MB
    unsigned short* Vtb = (unsigned short*)(ws + 56 * MB);  // 16 MB
    unsigned short* Ctx = (unsigned short*)(ws + 72 * MB);  // 16 MB

    // 1) pack to bf16
    const int nX4 = (4 * 2048 * 1024) / 4;   // 2097152
    const int nW4 = (1024 * 1024) / 4;       // 262144
    cvt_bf16_kernel<<<nX4 / 256, 256, 0, stream>>>(x,  Xbf, nX4);
    cvt_bf16_kernel<<<nW4 / 256, 256, 0, stream>>>(wq, Wqb, nW4);
    cvt_bf16_kernel<<<nW4 / 256, 256, 0, stream>>>(wk, Wkb, nW4);
    cvt_bf16_kernel<<<nW4 / 256, 256, 0, stream>>>(wv, Wvb, nW4);
    cvt_bf16_kernel<<<nW4 / 256, 256, 0, stream>>>(wo, Wob, nW4);

    // 2) projections (4096 waves each = 1024 blocks x 128 threads)
    const float qscale = 0.08838834764831845f;   // 1/sqrt(128)
    proj_wmma_kernel<<<1024, 128, 0, stream>>>(Xbf, Wqb, Qb,  nullptr, 0, qscale);
    proj_wmma_kernel<<<1024, 128, 0, stream>>>(Xbf, Wkb, Kb,  nullptr, 0, 1.0f);
    proj_wmma_kernel<<<1024, 128, 0, stream>>>(Xbf, Wvb, Vtb, nullptr, 1, 1.0f);

    // 3) flash attention (1024 blocks x 4 waves; 4 q-tiles per block share K/V)
    flash_attn_kernel<<<1024, 128, 0, stream>>>(Qb, Kb, Vtb, mask, Ctx);

    // 4) output projection -> f32
    proj_wmma_kernel<<<1024, 128, 0, stream>>>(Ctx, Wob, nullptr, out, 2, 1.0f);
}
